// PixelEncoder_88845693485536
// MI455X (gfx1250) — matrix operand order repaired and checked by reading, not compiled, not run on hardware
//
#include <hip/hip_runtime.h>
#include <hip/hip_bf16.h>

// ---------------------------------------------------------------------------
// PixelEncoder on gfx1250: fourier-feature encode + 5-layer ELU MLP (H=16).
// - One wave32 processes TWO 16-row tiles per iteration (independent WMMA
//   chains fill the 4-coexec WMMA->VALU hazard slots).
// - x rows are streamed global->LDS with GLOBAL_LOAD_ASYNC_TO_LDS_B128,
//   double-buffered and pipelined one iteration ahead (ASYNCcnt).
// - All matmuls run on v_wmma_f32_16x16x32_f16 (fp16 in, fp32 accumulate).
// ---------------------------------------------------------------------------

typedef __attribute__((ext_vector_type(16))) _Float16 v16h;
typedef __attribute__((ext_vector_type(8)))  _Float16 v8h;
typedef __attribute__((ext_vector_type(8)))  float    v8f;

#define PE_TWO_PI 6.283185f

static __device__ __forceinline__ void lds_fence() {
    // DS ops from one wave are serviced in-order by the LDS; the wait +
    // memory clobber stops the compiler reordering store->load exchanges.
    asm volatile("s_wait_dscnt 0" ::: "memory");
}
static __device__ __forceinline__ void async_wait_all() {
    asm volatile("s_wait_asynccnt 0" ::: "memory");
}
static __device__ __forceinline__ void async_wait_keep3() {
    // Current pair = first 3 async ops in flight; next pair = 3 more.
    // Async loads complete in order, so count<=3 => current pair landed.
    asm volatile("s_wait_asynccnt 3" ::: "memory");
}
// Async DMA: 16 bytes global -> LDS. GVS addressing: SGPR64 base + VGPR byte
// offset; VDST VGPR carries the wave-relative LDS byte address.
static __device__ __forceinline__ void async_b128(unsigned lds_off,
                                                  const float* sbase, int voff) {
    asm volatile("global_load_async_to_lds_b128 %0, %1, %2"
                 :
                 : "v"(lds_off), "v"(voff), "s"(sbase)
                 : "memory");
}

// Issue one tile-pair (2 tiles * 176 floats = 1408 B = 88 x b128 chunks).
static __device__ __forceinline__ void issue_pair_load(const float* __restrict__ x,
                                                       long pair, unsigned ldsbase,
                                                       int lane) {
    const float* src = x + pair * 352;          // 1408-byte aligned region
    int c0 = lane;                               // chunks 0..31
    async_b128(ldsbase + c0 * 16, src, c0 * 16);
    int c1 = lane + 32;                          // chunks 32..63
    async_b128(ldsbase + c1 * 16, src, c1 * 16);
    int c2 = lane + 64;                          // chunks 64..87
    if (c2 < 88) async_b128(ldsbase + c2 * 16, src, c2 * 16);
}

// Load a 16x32 fp16 A fragment (row = lane%16) from an LDS row-major tile.
// A layout (contiguous, matches B): lanes 0-15 hold K = kbase+{0..15},
// lanes 16-31 hold K = kbase+{16..31}.
static __device__ __forceinline__ v16h load_a_frag(const _Float16* base, int row,
                                                   int ko, int rowStride, int kbase) {
    const _Float16* r = base + row * rowStride + kbase;
    v8h lo = *(const v8h*)(r + 2 * ko);
    v8h hi = *(const v8h*)(r + 2 * ko + 8);
    return __builtin_shufflevector(lo, hi, 0,1,2,3,4,5,6,7,8,9,10,11,12,13,14,15);
}

static __device__ __forceinline__ v8f wmma_f16(v16h a, v16h b, v8f c) {
    // 8 args: (neg_a, A, neg_b, B, c_mod, C, reuse_a, reuse_b)
    return __builtin_amdgcn_wmma_f32_16x16x32_f16(false, a, false, b,
                                                  (short)0, c, false, false);
}

static __device__ __forceinline__ float elu_f(float v) {
    return v > 0.0f ? v : (__expf(v) - 1.0f);
}

__global__ void __launch_bounds__(256)
PixelEncoder_88845693485536_kernel(
    const float* __restrict__ x,  const float* __restrict__ Blight,
    const float* __restrict__ W1, const float* __restrict__ b1,
    const float* __restrict__ W2, const float* __restrict__ b2,
    const float* __restrict__ W3, const float* __restrict__ b3,
    const float* __restrict__ W4, const float* __restrict__ b4,
    const float* __restrict__ W5, const float* __restrict__ b5,
    float* __restrict__ out, int N, int tiles, int totalWaves)
{
    // ---- shared weights (fp16, transposed so B fragments are contiguous) ----
    // sWxT[col*K + k] = W[k][col]; K padded with zeros.
    __shared__ __align__(32) _Float16 sW1T[16 * 64];       // 41 -> 64
    __shared__ __align__(32) _Float16 sW2T[16 * 32];       // 16 -> 32
    __shared__ __align__(32) _Float16 sW3T[16 * 32];
    __shared__ __align__(32) _Float16 sW4T[16 * 32];
    __shared__ __align__(32) _Float16 sW5T[16 * 32];       // only col 0 nonzero
    __shared__ float sB[32];                               // B_light (2x16)
    // ---- per-wave scratch (8 waves / block, 2 tiles per iteration) ----
    __shared__ __align__(32) _Float16 sFeat[8][2][16 * 64]; // feature tiles
    __shared__ __align__(32) _Float16 sH[8][2][16 * 32];    // hidden tiles
    __shared__ __align__(16) float    sX[8][2][352];        // x, double-buffered

    const int tid  = threadIdx.x;
    const int wave = tid >> 5;
    const int lane = tid & 31;
    const int row  = lane & 15;        // A-layout row this lane owns
    const int col  = lane & 15;        // C-layout column this lane owns
    const int hi   = lane >> 4;
    const int ko   = hi ? 8 : 0;       // A-frag K offset
    const int k0   = hi ? 16 : 0;      // B-frag K offset
    const int m0   = hi ? 8 : 0;       // C-layout row base

    // ---- one-time: convert & pad weights into LDS -------------------------
    for (int i = tid; i < 16 * 64; i += 256) {
        int c = i >> 6, k = i & 63;
        sW1T[i] = (k < 41) ? (_Float16)W1[k * 16 + c] : (_Float16)0.0f;
    }
    for (int i = tid; i < 16 * 32; i += 256) {
        int c = i >> 5, k = i & 31;
        sW2T[i] = (k < 16) ? (_Float16)W2[k * 16 + c] : (_Float16)0.0f;
        sW3T[i] = (k < 16) ? (_Float16)W3[k * 16 + c] : (_Float16)0.0f;
        sW4T[i] = (k < 16) ? (_Float16)W4[k * 16 + c] : (_Float16)0.0f;
        sW5T[i] = (c == 0 && k < 16) ? (_Float16)W5[k] : (_Float16)0.0f;
    }
    if (tid < 32) sB[tid] = Blight[tid];
    // zero per-wave tiles once: padding columns stay zero forever
    for (int q = 0; q < 2; ++q) {
        for (int i = lane; i < 16 * 64; i += 32) sFeat[wave][q][i] = (_Float16)0.0f;
        for (int i = lane; i < 16 * 32; i += 32) sH[wave][q][i]    = (_Float16)0.0f;
    }
    __syncthreads();

    // ---- per-lane constants ----
    const float bias1 = b1[col];
    const float bias5 = b5[0];
    float biasMid[3] = { b2[col], b3[col], b4[col] };

    // B fragments live in registers for the whole kernel.
    // B layout: lane holds column N=lane%16; lanes 0-15 K=k..k+15, 16-31 K=+16.
    const v16h bw1_0 = *(const v16h*)(sW1T + col * 64 +  0 + k0);
    const v16h bw1_1 = *(const v16h*)(sW1T + col * 64 + 32 + k0);
    v16h bwMid[3];
    bwMid[0] = *(const v16h*)(sW2T + col * 32 + k0);
    bwMid[1] = *(const v16h*)(sW3T + col * 32 + k0);
    bwMid[2] = *(const v16h*)(sW4T + col * 32 + k0);
    const v16h bw5 = *(const v16h*)(sW5T + col * 32 + k0);

    unsigned xoff[2];
    xoff[0] = (unsigned)(size_t)(void*)&sX[wave][0][0];   // low 32 bits = LDS byte addr
    xoff[1] = (unsigned)(size_t)(void*)&sX[wave][1][0];

    const long  N11       = (long)N * 11;
    const long  npairs    = (long)(tiles + 1) >> 1;       // pair p = tiles 2p, 2p+1
    const long  waveId    = (long)blockIdx.x * 8 + wave;
    const v8f   zacc      = {0.f, 0.f, 0.f, 0.f, 0.f, 0.f, 0.f, 0.f};

    // ---- pipeline prologue: async-issue first pair ----
    int bufi = 0;
    if (waveId < npairs && (waveId + 1) * 352 <= N11)
        issue_pair_load(x, waveId, xoff[0], lane);

    for (long p = waveId; p < npairs; p += totalWaves, bufi ^= 1) {
        float* xcur = &sX[wave][bufi][0];
        const long pn     = p + totalWaves;
        const bool fast_p = ((p + 1) * 352 <= N11);
        const bool fast_n = (pn < npairs) && ((pn + 1) * 352 <= N11);

        // -- issue next pair's async DMA, then wait only for the current one --
        if (fast_n) issue_pair_load(x, pn, xoff[bufi ^ 1], lane);
        if (fast_p) {
            if (fast_n) async_wait_keep3();
            else        async_wait_all();
        } else {
            // tail pair: guarded scalar staging
            for (int i = lane; i < 352; i += 32) {
                long g = p * 352 + i;
                xcur[i] = (g < N11) ? x[g] : 0.0f;
            }
            lds_fence();
        }

        // warm GL2 two pairs ahead (global_prefetch_b8)
        long p2 = p + 2L * totalWaves;
        if (p2 < npairs && lane < 11)
            __builtin_prefetch(x + p2 * 352 + lane * 32, 0, 1);

        // -- fourier features for both sub-tiles (VALU/trans work) --
        #pragma unroll
        for (int q = 0; q < 2; ++q) {
            const float* xt = xcur + q * 176;
            _Float16*    ft = &sFeat[wave][q][0];
            const float lx = xt[row * 11 + 9];
            const float ly = xt[row * 11 + 10];
            if (!hi) {
                #pragma unroll
                for (int c = 0; c < 9; ++c)
                    ft[row * 64 + c] = (_Float16)xt[row * 11 + c];
                #pragma unroll
                for (int f = 0; f < 16; ++f) {
                    float ang = PE_TWO_PI * (lx * sB[f] + ly * sB[16 + f]);
                    ft[row * 64 + 9 + f] = (_Float16)__cosf(ang);
                }
            } else {
                #pragma unroll
                for (int f = 0; f < 16; ++f) {
                    float ang = PE_TWO_PI * (lx * sB[f] + ly * sB[16 + f]);
                    ft[row * 64 + 25 + f] = (_Float16)__sinf(ang);
                }
            }
        }
        lds_fence();

        // -- layer 1: 16x64 feats @ 64x16 W1 (two K=32 WMMAs per sub-tile) --
        v8f acc[2];
        #pragma unroll
        for (int q = 0; q < 2; ++q) {
            v8f a = wmma_f16(load_a_frag(&sFeat[wave][q][0], row, ko, 64, 0),  bw1_0, zacc);
            acc[q] = wmma_f16(load_a_frag(&sFeat[wave][q][0], row, ko, 64, 32), bw1_1, a);
        }
        #pragma unroll
        for (int q = 0; q < 2; ++q) {
            #pragma unroll
            for (int j = 0; j < 8; ++j) {
                float v = elu_f(acc[q][j] + bias1);
                sH[wave][q][(m0 + j) * 32 + col] = (_Float16)v;  // D->A transpose
            }
        }
        lds_fence();

        // -- layers 2..4: 16x16 @ 16x16 (K padded to 32 with persistent zeros) --
        #pragma unroll
        for (int L = 0; L < 3; ++L) {
            #pragma unroll
            for (int q = 0; q < 2; ++q)
                acc[q] = wmma_f16(load_a_frag(&sH[wave][q][0], row, ko, 32, 0),
                                  bwMid[L], zacc);
            #pragma unroll
            for (int q = 0; q < 2; ++q) {
                #pragma unroll
                for (int j = 0; j < 8; ++j) {
                    float v = elu_f(acc[q][j] + biasMid[L]);
                    sH[wave][q][(m0 + j) * 32 + col] = (_Float16)v;
                }
            }
            lds_fence();
        }

        // -- layer 5: h @ W5 (W5 packed into column 0 of B) --
        #pragma unroll
        for (int q = 0; q < 2; ++q)
            acc[q] = wmma_f16(load_a_frag(&sH[wave][q][0], row, ko, 32, 0), bw5, zacc);
        #pragma unroll
        for (int q = 0; q < 2; ++q) {
            if (col == 0) {
                long tb = (2 * p + q) * 16;
                #pragma unroll
                for (int j = 0; j < 8; ++j) {
                    long r = tb + m0 + j;
                    if (r < N) out[r] = acc[q][j] + bias5;
                }
            }
        }
        lds_fence();   // protect hbuf/feats before next iteration rewrites them
    }
}

extern "C" void kernel_launch(void* const* d_in, const int* in_sizes, int n_in,
                              void* d_out, int out_size, void* d_ws, size_t ws_size,
                              hipStream_t stream) {
    (void)n_in; (void)d_ws; (void)ws_size; (void)out_size;
    const float* x  = (const float*)d_in[0];
    const float* Bl = (const float*)d_in[1];
    const float* W1 = (const float*)d_in[2];
    const float* b1 = (const float*)d_in[3];
    const float* W2 = (const float*)d_in[4];
    const float* b2 = (const float*)d_in[5];
    const float* W3 = (const float*)d_in[6];
    const float* b3 = (const float*)d_in[7];
    const float* W4 = (const float*)d_in[8];
    const float* b4 = (const float*)d_in[9];
    const float* W5 = (const float*)d_in[10];
    const float* b5 = (const float*)d_in[11];
    float* out = (float*)d_out;

    const int N      = in_sizes[0] / 11;
    const int tiles  = (N + 15) / 16;          // 524288 for N = 8.39M
    const int npairs = (tiles + 1) / 2;
    int blocks = (npairs + 7) / 8;
    if (blocks > 2048) blocks = 2048;          // 16K waves, 2 tiles/iteration
    const int totalWaves = blocks * 8;

    PixelEncoder_88845693485536_kernel<<<blocks, 256, 0, stream>>>(
        x, Bl, W1, b1, W2, b2, W3, b3, W4, b4, W5, b5,
        out, N, tiles, totalWaves);
}
